// CustomGraphNet_81355270521078
// MI455X (gfx1250) — matrix operand order repaired and checked by reading, not compiled, not run on hardware
//
#include <hip/hip_runtime.h>
#include <hip/hip_fp16.h>

typedef __attribute__((ext_vector_type(16))) _Float16 v16h;
typedef __attribute__((ext_vector_type(8)))  _Float16 v8h;
typedef __attribute__((ext_vector_type(8)))  float    v8f;
typedef __attribute__((ext_vector_type(4)))  int      v4i;

#define NN 20000
#define NE 320000
#define LDIM 128

// ---- gfx1250 async Global->LDS path (guarded; falls back to VGPR copy) ----
#if defined(__has_builtin)
#if __has_builtin(__builtin_amdgcn_global_load_async_to_lds_b128) && \
    __has_builtin(__builtin_amdgcn_s_wait_asynccnt)
#define USE_ASYNC_LDS 1
#endif
#endif
#ifndef USE_ASYNC_LDS
#define USE_ASYNC_LDS 0
#endif

#if USE_ASYNC_LDS
typedef __attribute__((address_space(1))) v4i gas_v4i;
typedef __attribute__((address_space(3))) v4i las_v4i;
#endif

__device__ inline void cp16_g2l(const void* gsrc, void* ldst) {
#if USE_ASYNC_LDS
  // Per-lane async copy of 16 bytes, tracked by ASYNCcnt; LDS byte offset is
  // the low 32 bits of the generic shared pointer.
  __builtin_amdgcn_global_load_async_to_lds_b128(
      (gas_v4i*)(uintptr_t)gsrc,
      (las_v4i*)(unsigned)(uintptr_t)ldst,
      0, 0);
#else
  *(v8h*)ldst = *(const v8h*)gsrc;
#endif
}
__device__ inline void cp_join() {
#if USE_ASYNC_LDS
  __builtin_amdgcn_s_wait_asynccnt(0);
#endif
}

// ---------------- WMMA GEMM core ----------------
// One wave computes a 16-row x 128-col tile: D = A(16xKP) @ Wt^T + bias.
// A panel lives in LDS (f16, row-major, stride aStride halves).
// Wt is the transposed weight in global: Wt[n][k], n-major, KP halves per row.
enum { EPI_RELU = 0, EPI_ADD_RELU = 1, EPI_F32 = 2 };

struct MlpW {
  const _Float16 *wt_in, *wt_r0a, *wt_r0b, *wt_r1a, *wt_r1b, *wt_out;
  const float *b_in, *b_r0a, *b_r0b, *b_r1a, *b_r1b, *b_out, *gamma, *beta;
};

template <int KP>
__device__ inline void gemm16(const _Float16* __restrict__ Ap, int aStride,
                              const _Float16* __restrict__ Wt,
                              const float* __restrict__ bias,
                              _Float16* __restrict__ Out,
                              float* __restrict__ OutF, int epi) {
  const int lane = threadIdx.x & 31;
  const int half = lane >> 4;   // K sub-block select (lanes 16-31 carry K+8 / K+24)
  const int nl   = lane & 15;   // row (A) / column (B,C)

  v8f acc[8];
#pragma unroll
  for (int nt = 0; nt < 8; ++nt) {
    float b = bias[nt * 16 + nl];
#pragma unroll
    for (int i = 0; i < 8; ++i) acc[nt][i] = b;
  }

#pragma unroll
  for (int kc = 0; kc < KP / 32; ++kc) {
    // A fragment: lane nl holds row nl; K pieces [kc*32+half*8 .. +7] and [+16..].
    const _Float16* pa = Ap + nl * aStride + kc * 32 + half * 8;
    v8h a0 = *(const v8h*)pa;
    v8h a1 = *(const v8h*)(pa + 16);
    v16h a;
#pragma unroll
    for (int i = 0; i < 8; ++i) { a[i] = a0[i]; a[i + 8] = a1[i]; }

    if (kc + 1 < KP / 32)
      __builtin_prefetch(Wt + (size_t)nl * KP + (kc + 1) * 32, 0, 3);

#pragma unroll
    for (int nt = 0; nt < 8; ++nt) {
      // B fragment: lane holds column nt*16+nl; K pieces [kc*32+half*8 .. +7] and [+16..].
      const _Float16* pb = Wt + (size_t)(nt * 16 + nl) * KP + kc * 32 + half * 8;
      v8h b0 = *(const v8h*)pb;
      v8h b1 = *(const v8h*)(pb + 16);
      v16h bf;
#pragma unroll
      for (int i = 0; i < 8; ++i) { bf[i] = b0[i]; bf[i + 8] = b1[i]; }
      acc[nt] = __builtin_amdgcn_wmma_f32_16x16x32_f16(
          false, a, false, bf, (short)0, acc[nt], false, false);
    }
  }

  // Epilogue: C tile layout -> VGPR v holds row v (+8 for lanes 16-31), col nl.
#pragma unroll
  for (int nt = 0; nt < 8; ++nt) {
#pragma unroll
    for (int v = 0; v < 8; ++v) {
      int m = v + half * 8;
      int n = nt * 16 + nl;
      float x = acc[nt][v];
      if (epi == EPI_RELU)          Out[m * LDIM + n] = (_Float16)fmaxf(x, 0.f);
      else if (epi == EPI_ADD_RELU) Out[m * LDIM + n] =
          (_Float16)((float)Out[m * LDIM + n] + fmaxf(x, 0.f));
      else                          OutF[m * LDIM + n] = x;
    }
  }
}

// in -> relu -> 2 residual blocks; result (f16) left in H.
template <int KP>
__device__ inline void mlp_core(const _Float16* A, int aStride, _Float16* H,
                                _Float16* T, const MlpW& w) {
  gemm16<KP>(A, aStride, w.wt_in, w.b_in, H, nullptr, EPI_RELU);
  __syncthreads();
  gemm16<LDIM>(H, LDIM, w.wt_r0a, w.b_r0a, T, nullptr, EPI_RELU);
  __syncthreads();
  gemm16<LDIM>(T, LDIM, w.wt_r0b, w.b_r0b, H, nullptr, EPI_ADD_RELU);
  __syncthreads();
  gemm16<LDIM>(H, LDIM, w.wt_r1a, w.b_r1a, T, nullptr, EPI_RELU);
  __syncthreads();
  gemm16<LDIM>(T, LDIM, w.wt_r1b, w.b_r1b, H, nullptr, EPI_ADD_RELU);
  __syncthreads();
}

// Final linear + LayerNorm (+ optional skip add from outG) -> f16 global.
__device__ inline void out_ln(const _Float16* H, float* F, const MlpW& w,
                              _Float16* outG, long row0, bool skipAdd) {
  gemm16<LDIM>(H, LDIM, w.wt_out, w.b_out, nullptr, F, EPI_F32);
  __syncthreads();
  const int lane = threadIdx.x & 31;
  const int m = lane & 15;   // row
  const int hh = lane >> 4;  // which half of the 128 columns
  const float* Fr = F + m * LDIM;
  float s = 0.f;
  for (int n = hh * 64; n < hh * 64 + 64; ++n) s += Fr[n];
  s += __shfl_xor(s, 16, 32);
  float mu = s * (1.0f / 128.0f);
  float vv = 0.f;
  for (int n = hh * 64; n < hh * 64 + 64; ++n) { float d = Fr[n] - mu; vv += d * d; }
  vv += __shfl_xor(vv, 16, 32);
  float rstd = rsqrtf(vv * (1.0f / 128.0f) + 1e-5f);
  _Float16* orow = outG + (size_t)(row0 + m) * LDIM;
  for (int n = hh * 64; n < hh * 64 + 64; ++n) {
    float y = (Fr[n] - mu) * rstd * w.gamma[n] + w.beta[n];
    if (skipAdd) y += (float)orow[n];
    orow[n] = (_Float16)y;
  }
}

// ---------------- Kernels ----------------

// Encoder: f32 input [rows x IN] -> f16 latent [rows x 128].
template <int IN>
__global__ void __launch_bounds__(64) enc_kernel(const float* __restrict__ in,
                                                 MlpW w,
                                                 _Float16* __restrict__ outG) {
  __shared__ __align__(16) _Float16 sA[2][16 * 256];  // also reused as f32 F
  __shared__ __align__(16) _Float16 sH[2][16 * LDIM];
  __shared__ __align__(16) _Float16 sT[2][16 * LDIM];
  const int wid = threadIdx.x >> 5, lane = threadIdx.x & 31;
  const long row0 = ((long)blockIdx.x * 2 + wid) * 16;
  _Float16 *A = sA[wid], *H = sH[wid], *T = sT[wid];
  for (int t = lane; t < 16 * 32; t += 32) {
    int m = t >> 5, k = t & 31;
    float x = (k < IN) ? in[(size_t)(row0 + m) * IN + k] : 0.f;
    A[m * 256 + k] = (_Float16)x;
  }
  __syncthreads();
  mlp_core<32>(A, 256, H, T, w);
  out_ln(H, (float*)A, w, outG, row0, false);
}

// Processor edge MLP: gather [lat[dst] | lat[src] | edge_lat] (384) -> new_e (in place).
__global__ void __launch_bounds__(64) edge_kernel(const _Float16* __restrict__ nodeLat,
                                                  _Float16* __restrict__ edgeLat,
                                                  const int* __restrict__ srcIdx,
                                                  const int* __restrict__ dstIdx,
                                                  MlpW w) {
  __shared__ __align__(16) _Float16 sA[2][16 * 384];  // reused as f32 F
  __shared__ __align__(16) _Float16 sH[2][16 * LDIM];
  __shared__ __align__(16) _Float16 sT[2][16 * LDIM];
  __shared__ int sI[2][2][16];
  const int wid = threadIdx.x >> 5, lane = threadIdx.x & 31;
  const long row0 = ((long)blockIdx.x * 2 + wid) * 16;
  if (lane < 16) {
    sI[wid][0][lane] = dstIdx[row0 + lane];
    sI[wid][1][lane] = srcIdx[row0 + lane];
  }
  __syncthreads();
  _Float16 *A = sA[wid], *H = sH[wid], *T = sT[wid];
  for (int t = lane; t < 16 * 48; t += 32) {
    int m = t / 48, c = t % 48, k = c * 8;
    const _Float16* p;
    if (k < 128)      p = nodeLat + (size_t)sI[wid][0][m] * LDIM + k;
    else if (k < 256) p = nodeLat + (size_t)sI[wid][1][m] * LDIM + (k - 128);
    else              p = edgeLat + (size_t)(row0 + m) * LDIM + (k - 256);
    cp16_g2l(p, A + m * 384 + k);
  }
  cp_join();
  __syncthreads();
  mlp_core<384>(A, 384, H, T, w);
  out_ln(H, (float*)A, w, edgeLat, row0, true);  // new_e = old_e + LN(mlp)
}

// Processor node MLP: [node_lat | agg] (256) -> new node lat (in place).
__global__ void __launch_bounds__(64) node_kernel(_Float16* __restrict__ nodeLat,
                                                  const float* __restrict__ agg,
                                                  MlpW w) {
  __shared__ __align__(16) _Float16 sA[2][16 * 256];
  __shared__ __align__(16) _Float16 sH[2][16 * LDIM];
  __shared__ __align__(16) _Float16 sT[2][16 * LDIM];
  const int wid = threadIdx.x >> 5, lane = threadIdx.x & 31;
  const long row0 = ((long)blockIdx.x * 2 + wid) * 16;
  _Float16 *A = sA[wid], *H = sH[wid], *T = sT[wid];
  for (int t = lane; t < 16 * 32; t += 32) {
    int m = t >> 5, k = (t & 31) * 8;
    if (k < 128) {
      cp16_g2l(nodeLat + (size_t)(row0 + m) * LDIM + k, A + m * 256 + k);
    } else {
      const float* q = agg + (size_t)(row0 + m) * LDIM + (k - 128);
      v8h h;
#pragma unroll
      for (int i = 0; i < 8; ++i) h[i] = (_Float16)q[i];
      *(v8h*)(A + m * 256 + k) = h;
    }
  }
  cp_join();
  __syncthreads();
  mlp_core<256>(A, 256, H, T, w);
  out_ln(H, (float*)A, w, nodeLat, row0, true);
}

// Decoder: node lat -> [rows x 3] f32 (no LN; tiny W_out done scalar).
__global__ void __launch_bounds__(64) dec_kernel(const _Float16* __restrict__ nodeLat,
                                                 MlpW w,
                                                 const float* __restrict__ wout,
                                                 const float* __restrict__ bout,
                                                 float* __restrict__ out) {
  __shared__ __align__(16) _Float16 sA[2][16 * LDIM];
  __shared__ __align__(16) _Float16 sH[2][16 * LDIM];
  __shared__ __align__(16) _Float16 sT[2][16 * LDIM];
  const int wid = threadIdx.x >> 5, lane = threadIdx.x & 31;
  const long row0 = ((long)blockIdx.x * 2 + wid) * 16;
  _Float16 *A = sA[wid], *H = sH[wid], *T = sT[wid];
  for (int t = lane; t < 16 * 16; t += 32) {
    int m = t >> 4, k = (t & 15) * 8;
    cp16_g2l(nodeLat + (size_t)(row0 + m) * LDIM + k, A + m * LDIM + k);
  }
  cp_join();
  __syncthreads();
  mlp_core<LDIM>(A, LDIM, H, T, w);
  for (int j = lane; j < 48; j += 32) {
    int m = j / 3, o = j % 3;
    float s = bout[o];
    for (int k = 0; k < LDIM; ++k) s += (float)H[m * LDIM + k] * wout[k * 3 + o];
    out[(size_t)(row0 + m) * 3 + o] = s;
  }
}

// ---------------- Attention / segment softmax ----------------
__device__ inline unsigned enc_ord(float f) {
  unsigned u = __float_as_uint(f);
  return (u & 0x80000000u) ? ~u : (u | 0x80000000u);
}
__device__ inline float dec_ord(unsigned u) {
  unsigned v = (u & 0x80000000u) ? (u & 0x7fffffffu) : ~u;
  return __uint_as_float(v);
}

__global__ void att_score_kernel(const _Float16* __restrict__ E,
                                 const int* __restrict__ dst,
                                 const float* __restrict__ W1,
                                 const float* __restrict__ b1,
                                 const float* __restrict__ W2,
                                 float* __restrict__ score,
                                 unsigned* __restrict__ mx) {
  int e = blockIdx.x * blockDim.x + threadIdx.x;
  if (e >= NE) return;
  const _Float16* r = E + (size_t)e * LDIM;
  float h0 = b1[0], h1 = b1[1], h2 = b1[2], h3 = b1[3];
  for (int k = 0; k < LDIM; ++k) {
    float x = (float)r[k];
    h0 += x * W1[k * 4 + 0]; h1 += x * W1[k * 4 + 1];
    h2 += x * W1[k * 4 + 2]; h3 += x * W1[k * 4 + 3];
  }
  float s = fmaxf(h0, 0.f) * W2[0] + fmaxf(h1, 0.f) * W2[1] +
            fmaxf(h2, 0.f) * W2[2] + fmaxf(h3, 0.f) * W2[3];
  score[e] = s;
  atomicMax(&mx[dst[e]], enc_ord(s));
}

__global__ void att_exp_kernel(const float* __restrict__ score,
                               const int* __restrict__ dst,
                               const unsigned* __restrict__ mx,
                               float* __restrict__ escore,
                               float* __restrict__ denom) {
  int e = blockIdx.x * blockDim.x + threadIdx.x;
  if (e >= NE) return;
  int d = dst[e];
  float ee = __expf(score[e] - dec_ord(mx[d]));
  escore[e] = ee;
  atomicAdd(&denom[d], ee);
}

__global__ void att_agg_kernel(const _Float16* __restrict__ E,
                               const int* __restrict__ dst,
                               const float* __restrict__ escore,
                               const float* __restrict__ denom,
                               float* __restrict__ agg) {
  int t = blockIdx.x * blockDim.x + threadIdx.x;
  int e = t >> 5, lane = t & 31;
  if (e >= NE) return;
  int d = dst[e];
  float alpha = escore[e] / denom[d];
  const _Float16* r = E + (size_t)e * LDIM;
  for (int c = lane * 4; c < lane * 4 + 4; ++c)
    atomicAdd(&agg[(size_t)d * LDIM + c], alpha * (float)r[c]);
}

// ---------------- Utility kernels ----------------
__global__ void prep_w_kernel(const float* __restrict__ w, _Float16* __restrict__ wt,
                              int K, int N, int KP) {
  int i = blockIdx.x * blockDim.x + threadIdx.x;
  if (i >= N * KP) return;
  int n = i / KP, k = i % KP;
  wt[i] = (k < K) ? (_Float16)w[(size_t)k * N + n] : (_Float16)0.f;
}

__global__ void zero_f32_kernel(float* p, int n) {
  int i = blockIdx.x * blockDim.x + threadIdx.x;
  if (i < n) p[i] = 0.f;
}

// ---------------- Host launcher ----------------
extern "C" void kernel_launch(void* const* d_in, const int* in_sizes, int n_in,
                              void* d_out, int out_size, void* d_ws, size_t ws_size,
                              hipStream_t stream) {
  (void)in_sizes; (void)n_in; (void)out_size; (void)ws_size;
  char* wsBase = (char*)d_ws;
  size_t cur = 0;
  auto alloc = [&](size_t bytes) -> void* {
    cur = (cur + 255) & ~(size_t)255;
    void* p = wsBase + cur;
    cur += bytes;
    return p;
  };
  auto prep = [&](const void* w, int K, int N, int KP) -> const _Float16* {
    _Float16* dst = (_Float16*)alloc((size_t)N * KP * sizeof(_Float16));
    int total = N * KP;
    prep_w_kernel<<<(total + 255) / 256, 256, 0, stream>>>((const float*)w, dst, K, N, KP);
    return dst;
  };
  auto mkMlp = [&](int base, bool ln, int inDim, int KP, bool prepOut) -> MlpW {
    MlpW m{};
    m.wt_in  = prep(d_in[base + 0], inDim, LDIM, KP);  m.b_in  = (const float*)d_in[base + 1];
    m.wt_r0a = prep(d_in[base + 2], LDIM, LDIM, LDIM); m.b_r0a = (const float*)d_in[base + 3];
    m.wt_r0b = prep(d_in[base + 4], LDIM, LDIM, LDIM); m.b_r0b = (const float*)d_in[base + 5];
    m.wt_r1a = prep(d_in[base + 6], LDIM, LDIM, LDIM); m.b_r1a = (const float*)d_in[base + 7];
    m.wt_r1b = prep(d_in[base + 8], LDIM, LDIM, LDIM); m.b_r1b = (const float*)d_in[base + 9];
    if (prepOut) m.wt_out = prep(d_in[base + 10], LDIM, LDIM, LDIM);
    m.b_out = (const float*)d_in[base + 11];
    if (ln) { m.gamma = (const float*)d_in[base + 12]; m.beta = (const float*)d_in[base + 13]; }
    return m;
  };

  // Input leaf indices (setup_inputs insertion order).
  const float* x         = (const float*)d_in[0];
  const float* edge_attr = (const float*)d_in[1];
  const int*   eidx      = (const int*)d_in[2];
  const int*   srcIdx    = eidx;        // edge_index[0]
  const int*   dstIdx    = eidx + NE;   // edge_index[1]

  const int B_NENC = 3, B_EENC = 17;
  const int B_P0E = 31, B_P0N = 45, B_P0A = 59;
  const int B_P1E = 62, B_P1N = 76, B_P1A = 90;
  const int B_DEC = 93;

  MlpW mNE = mkMlp(B_NENC, true, 16, 32, true);
  MlpW mEE = mkMlp(B_EENC, true, 8, 32, true);
  MlpW mPE[2] = { mkMlp(B_P0E, true, 3 * LDIM, 3 * LDIM, true),
                  mkMlp(B_P1E, true, 3 * LDIM, 3 * LDIM, true) };
  MlpW mPN[2] = { mkMlp(B_P0N, true, 2 * LDIM, 2 * LDIM, true),
                  mkMlp(B_P1N, true, 2 * LDIM, 2 * LDIM, true) };
  const float* attW1[2] = { (const float*)d_in[B_P0A], (const float*)d_in[B_P1A] };
  const float* attB1[2] = { (const float*)d_in[B_P0A + 1], (const float*)d_in[B_P1A + 1] };
  const float* attW2[2] = { (const float*)d_in[B_P0A + 2], (const float*)d_in[B_P1A + 2] };
  MlpW mDec = mkMlp(B_DEC, false, LDIM, LDIM, false);
  const float* decWout = (const float*)d_in[B_DEC + 10];
  const float* decBout = (const float*)d_in[B_DEC + 11];

  // Latents + attention scratch.
  _Float16* nodeLat = (_Float16*)alloc((size_t)NN * LDIM * sizeof(_Float16));
  _Float16* edgeLat = (_Float16*)alloc((size_t)NE * LDIM * sizeof(_Float16));
  float*    agg     = (float*)alloc((size_t)NN * LDIM * sizeof(float));
  float*    denom   = (float*)alloc((size_t)NN * sizeof(float));
  unsigned* mx      = (unsigned*)alloc((size_t)NN * sizeof(unsigned));
  float*    score   = (float*)alloc((size_t)NE * sizeof(float));
  float*    escore  = (float*)alloc((size_t)NE * sizeof(float));

  // Encoders.
  enc_kernel<16><<<NN / 32, 64, 0, stream>>>(x, mNE, nodeLat);
  enc_kernel<8><<<NE / 32, 64, 0, stream>>>(edge_attr, mEE, edgeLat);

  // Message passes.
  for (int p = 0; p < 2; ++p) {
    zero_f32_kernel<<<(NN * LDIM + 255) / 256, 256, 0, stream>>>(agg, NN * LDIM);
    zero_f32_kernel<<<(NN + 255) / 256, 256, 0, stream>>>(denom, NN);
    zero_f32_kernel<<<(NN + 255) / 256, 256, 0, stream>>>((float*)mx, NN);

    edge_kernel<<<NE / 32, 64, 0, stream>>>(nodeLat, edgeLat, srcIdx, dstIdx, mPE[p]);
    att_score_kernel<<<(NE + 255) / 256, 256, 0, stream>>>(
        edgeLat, dstIdx, attW1[p], attB1[p], attW2[p], score, mx);
    att_exp_kernel<<<(NE + 255) / 256, 256, 0, stream>>>(score, dstIdx, mx, escore, denom);
    att_agg_kernel<<<(NE * 32 + 255) / 256, 256, 0, stream>>>(
        edgeLat, dstIdx, escore, denom, agg);
    node_kernel<<<NN / 32, 64, 0, stream>>>(nodeLat, agg, mPN[p]);
  }

  // Decoder.
  dec_kernel<<<NN / 32, 64, 0, stream>>>(nodeLat, mDec, decWout, decBout, (float*)d_out);
}
